// ConexaoEsparsa_15753940042200
// MI455X (gfx1250) — matrix operand order repaired and checked by reading, not compiled, not run on hardware
//
#include <hip/hip_runtime.h>

typedef __attribute__((ext_vector_type(16))) _Float16 v16h;
typedef __attribute__((ext_vector_type(8)))  float    v8f;

#define B_DIM   256
#define IN_DIM  1024
#define OUT_DIM 512
#define DS_DIM  128          // D*S = 16*8
#define KC      32           // K chunk = one WMMA K-step
#define ODS     (OUT_DIM * DS_DIM)   // 65536, stride of b in output

// out[b,o,ds] (b-major) ; one block per o ; 8 waves, wave w owns ds-tile w,
// accumulates all 16 b-tiles (16 x v8f = 128 accumulator VGPRs per lane).
__global__ __launch_bounds__(256, 1)
void conexao_esparsa_wmma(const float* __restrict__ x,
                          const float* __restrict__ peso,
                          const float* __restrict__ mascara,
                          float* __restrict__ out)
{
    // f16 operand staging, pre-swizzled to WMMA lane layouts (32B/lane rows)
    __shared__ _Float16 __attribute__((aligned(32))) ldsA[16][32][16]; // 16 KB
    __shared__ _Float16 __attribute__((aligned(32))) ldsB[ 8][32][16]; //  8 KB
    __shared__ int s_list[IN_DIM];                                     //  4 KB
    __shared__ int s_cnt;

    const int o    = blockIdx.x;
    const int tid  = threadIdx.x;
    const int lane = tid & 31;
    const int wave = tid >> 5;

    // ---- compact non-zero mask rows for this o (order irrelevant) ----
    if (tid == 0) s_cnt = 0;
    __syncthreads();
    for (int i = tid; i < IN_DIM; i += 256) {
        if (mascara[(size_t)o * IN_DIM + i] != 0.0f) {
            int p = atomicAdd(&s_cnt, 1);
            s_list[p] = i;
        }
    }
    __syncthreads();
    const int cnt = s_cnt;
    const int nk  = (cnt + (KC - 1)) & ~(KC - 1);
    for (int idx = cnt + tid; idx < nk; idx += 256) s_list[idx] = -1; // pad
    __syncthreads();

    v8f acc[16] = {};   // zero-initialized accumulators

    for (int kc = 0; kc < nk; kc += KC) {
        // ---- stage A: gathered x columns -> f16, A-operand swizzle ----
        // A(m,k) element for b-tile t: x[b = 16t+m, i = list[kc+k]]
        #pragma unroll 4
        for (int j = 0; j < 32; ++j) {
            int e = tid + 256 * j;          // 0..8191 : (b, slot)
            int b = e >> 5;                 // 0..255
            int k = e & 31;                 // slot in chunk
            int ie = s_list[kc + k];
            float av = 0.0f;
            if (ie >= 0) av = x[(size_t)b * IN_DIM + ie];
            // inverse of ISA 16-bit A-layout (16x32): lane = g*16+m, half = 2v+p
            int m  = b & 15;
            int tl = b >> 4;
            int p  = k & 1;
            int kq = (k & 15) - p;                         // even 0..14
            int g  = (kq >= 8) ? 1 : 0;
            int vr = ((kq - 8 * g) >> 1) + ((k >= 16) ? 4 : 0);
            ldsA[tl][g * 16 + m][2 * vr + p] = (_Float16)av;
        }

        // ---- stage B: surviving peso rows -> f16, B-operand swizzle ----
        // B(k,n) for ds-tile t: peso[o, list[kc+k], 16t+n]  (mask==1 when kept)
        #pragma unroll
        for (int j = 0; j < 4; ++j) {
            int f   = tid + 256 * j;        // 0..1023 float4 ids
            int k   = f >> 5;               // slot 0..31
            int ds0 = (f & 31) << 2;        // 0,4,...,124
            int ie  = s_list[kc + k];
            float4 v = make_float4(0.0f, 0.0f, 0.0f, 0.0f);
            if (ie >= 0)
                v = *(const float4*)(peso + ((size_t)o * IN_DIM + ie) * DS_DIM + ds0);
            if (kc + KC < nk) {
                int ie2 = s_list[kc + KC + k];
                if (ie2 >= 0)
                    __builtin_prefetch(peso + ((size_t)o * IN_DIM + ie2) * DS_DIM + ds0, 0, 1);
            }
            // B layout (KxN=32x16 f16): lane = ((k>>3)&1)*16 + n, half = (k&7) + (k>=16)*8
            int lB = ((k >> 3) & 1) * 16;
            int h  = (k & 7) + ((k >= 16) ? 8 : 0);
            int tl = ds0 >> 4;
            int n0 = ds0 & 15;
            float vv[4] = {v.x, v.y, v.z, v.w};
            #pragma unroll
            for (int e2 = 0; e2 < 4; ++e2)
                ldsB[tl][lB + n0 + e2][h] = (_Float16)vv[e2];
        }

        __syncthreads();

        // ---- WMMA: wave owns ds-tile `wave`; sweep 16 b-tiles ----
        v16h bfrag = *(const v16h*)&ldsB[wave][lane][0];
        #pragma unroll
        for (int t = 0; t < 16; ++t) {
            v16h afrag = *(const v16h*)&ldsA[t][lane][0];
            acc[t] = __builtin_amdgcn_wmma_f32_16x16x32_f16(
                false, afrag, false, bfrag, (short)0, acc[t], false, false);
        }

        __syncthreads();
    }

    // ---- store D: C/D layout VGPR r -> (M = r + 8*(lane>=16), N = lane&15) ----
    const int g16 = lane >> 4;
    const int n   = lane & 15;
    const int dsg = wave * 16 + n;
    #pragma unroll
    for (int t = 0; t < 16; ++t) {
        #pragma unroll
        for (int r = 0; r < 8; ++r) {
            int b = t * 16 + r + 8 * g16;
            out[(size_t)b * ODS + (size_t)o * DS_DIM + dsg] = acc[t][r];
        }
    }
}

extern "C" void kernel_launch(void* const* d_in, const int* in_sizes, int n_in,
                              void* d_out, int out_size, void* d_ws, size_t ws_size,
                              hipStream_t stream) {
    (void)in_sizes; (void)n_in; (void)d_ws; (void)ws_size; (void)out_size;
    const float* x    = (const float*)d_in[0];
    const float* peso = (const float*)d_in[1];
    const float* masc = (const float*)d_in[2];
    float* out = (float*)d_out;
    conexao_esparsa_wmma<<<dim3(OUT_DIM), dim3(256), 0, stream>>>(x, peso, masc, out);
}
